// RoutingLayer_43611097923791
// MI455X (gfx1250) — matrix operand order repaired and checked by reading, not compiled, not run on hardware
//
#include <hip/hip_runtime.h>

typedef __attribute__((ext_vector_type(16))) __bf16 v16bf;
typedef __attribute__((ext_vector_type(8)))  float  v8f;

#define N_SAMP  16384
#define D_IN    1024
#define KD      4096
#define NCAP    64
#define DCAP    64
#define TM      128
#define TN      128
#define TK      64
#define LDSTB   72      // LDS row stride in bf16 elems (144B: 16B aligned, bank-rotating)

__device__ __forceinline__ unsigned short f2bf(float f) {
    unsigned u = __float_as_uint(f);
    unsigned r = u + 0x7FFFu + ((u >> 16) & 1u);   // round-to-nearest-even
    return (unsigned short)(r >> 16);
}
__device__ __forceinline__ float bf2f(unsigned short h) {
    return __uint_as_float(((unsigned)h) << 16);
}

union FragU { uint4 q[2]; v16bf v; };

// CDNA5 async global->LDS copy (ASYNCcnt-tracked), GV addressing mode.
// lds_off = wave-relative LDS byte address (low 32 bits of flat shared pointer).
__device__ __forceinline__ void async_load_b128(unsigned lds_off, const void* gaddr) {
    asm volatile("global_load_async_to_lds_b128 %0, %1, off"
                 :: "v"(lds_off), "v"(gaddr) : "memory");
}

// ---------------- fp32 -> bf16 conversions ----------------
__global__ __launch_bounds__(256) void kConvert(const float* __restrict__ src,
                                                unsigned short* __restrict__ dst, int n) {
    int i = blockIdx.x * blockDim.x + threadIdx.x;
    int stride = gridDim.x * blockDim.x;
    for (; i < n; i += stride) dst[i] = f2bf(src[i]);
}

// W (1024,4096) -> Wt (4096,1024) bf16, coalesced on the read side
__global__ __launch_bounds__(256) void kConvertWT(const float* __restrict__ W,
                                                  unsigned short* __restrict__ Wt) {
    int i = blockIdx.x * blockDim.x + threadIdx.x;
    int stride = gridDim.x * blockDim.x;
    const int total = KD * D_IN;
    for (; i < total; i += stride) {
        int n = i & (KD - 1);
        int k = i >> 12;
        Wt[(size_t)n * D_IN + k] = f2bf(W[(size_t)k * KD + n]);
    }
}

// ---------------- GEMM: u_hat = x @ W + b (bf16 in, bf16 out, f32 acc) ----------------
// Double-buffered LDS fed by async global->LDS B128 copies; 16 WMMAs per barrier pair.
__global__ __launch_bounds__(256) void kGemm(const unsigned short* __restrict__ xb,
                                             const unsigned short* __restrict__ wt,
                                             const float* __restrict__ bias,
                                             unsigned short* __restrict__ u) {
    __shared__ __align__(16) unsigned short Al[2][TM * LDSTB];
    __shared__ __align__(16) unsigned short Bl[2][TN * LDSTB];

    const int tid  = threadIdx.x;
    const int lane = tid & 31;
    const int wave = tid >> 5;
    const int wm   = wave >> 1;          // 0..3 -> 32-row strip
    const int wn   = wave & 1;           // 0..1 -> 64-col strip
    const int m0   = blockIdx.y * TM;
    const int n0   = blockIdx.x * TN;

    v8f acc[2][4];
    #pragma unroll
    for (int mt = 0; mt < 2; ++mt)
        #pragma unroll
        for (int nt = 0; nt < 4; ++nt) acc[mt][nt] = v8f{};

    // tile loading: thread -> (row, 64B half-row); 4 async B128 per matrix per tile
    const int lrow  = tid >> 1;          // 0..127
    const int lhalf = (tid & 1) * 32;    // elem offset 0 or 32 (=64B)

    const unsigned short* gA = xb + (size_t)(m0 + lrow) * D_IN + lhalf;
    const unsigned short* gB = wt + (size_t)(n0 + lrow) * D_IN + lhalf;
    unsigned ldsA[2], ldsB[2];
    #pragma unroll
    for (int pp = 0; pp < 2; ++pp) {
        ldsA[pp] = (unsigned)(size_t)&Al[pp][lrow * LDSTB + lhalf];
        ldsB[pp] = (unsigned)(size_t)&Bl[pp][lrow * LDSTB + lhalf];
    }

    #define ISSUE_TILE(buf, kk)                                             \
        {                                                                   \
            const unsigned short* a_ = gA + (kk);                           \
            const unsigned short* b_ = gB + (kk);                           \
            unsigned la_ = ldsA[buf], lb_ = ldsB[buf];                      \
            _Pragma("unroll")                                               \
            for (int i_ = 0; i_ < 4; ++i_) {                                \
                async_load_b128(la_ + i_ * 16, a_ + i_ * 8);                \
                async_load_b128(lb_ + i_ * 16, b_ + i_ * 8);                \
            }                                                               \
        }

    ISSUE_TILE(0, 0);
    int p = 0;
    for (int k0 = 0; k0 < D_IN; k0 += TK) {
        const bool more = (k0 + TK) < D_IN;
        if (more) {
            ISSUE_TILE(p ^ 1, k0 + TK);
            // 8 outstanding from current tile + 8 just issued; async loads
            // complete in order, so <=8 means the current tile is resident.
            asm volatile("s_wait_asynccnt 8" ::: "memory");
        } else {
            asm volatile("s_wait_asynccnt 0" ::: "memory");
        }
        __syncthreads();

        const unsigned short* Ab = &Al[p][0];
        const unsigned short* Bb = &Bl[p][0];
        #pragma unroll
        for (int ks = 0; ks < 2; ++ks) {
            const int kc = ks * 32;
            // A fragments: 16x32 bf16 (lane<16: K{0..7,16..23}; lane>=16: K{8..15,24..31})
            const int ac0 = kc + (lane >> 4) * 8;
            v16bf afrag[2];
            #pragma unroll
            for (int mt = 0; mt < 2; ++mt) {
                int row = wm * 32 + mt * 16 + (lane & 15);
                FragU f;
                f.q[0] = *(const uint4*)&Ab[row * LDSTB + ac0];
                f.q[1] = *(const uint4*)&Ab[row * LDSTB + ac0 + 16];
                afrag[mt] = f.v;
            }
            // B fragments: 32x16 bf16 (lane<16: K{0..7,16..23}; lane>=16: K{8..15,24..31})
            const int bc0 = kc + (lane >> 4) * 8;
            v16bf bfrag[4];
            #pragma unroll
            for (int nt = 0; nt < 4; ++nt) {
                int row = wn * 64 + nt * 16 + (lane & 15);
                FragU f;
                f.q[0] = *(const uint4*)&Bb[row * LDSTB + bc0];
                f.q[1] = *(const uint4*)&Bb[row * LDSTB + bc0 + 16];
                bfrag[nt] = f.v;
            }
            #pragma unroll
            for (int mt = 0; mt < 2; ++mt)
                #pragma unroll
                for (int nt = 0; nt < 4; ++nt)
                    acc[mt][nt] = __builtin_amdgcn_wmma_f32_16x16x32_bf16(
                        false, afrag[mt], false, bfrag[nt],
                        (short)0, acc[mt][nt], false, false);
        }
        __syncthreads();
        p ^= 1;
    }

    // epilogue: + bias, store bf16 u_hat
    #pragma unroll
    for (int mt = 0; mt < 2; ++mt) {
        #pragma unroll
        for (int nt = 0; nt < 4; ++nt) {
            int col   = n0 + wn * 64 + nt * 16 + (lane & 15);
            float bv  = bias[col];
            int rbase = m0 + wm * 32 + mt * 16 + ((lane >> 4) * 8);
            #pragma unroll
            for (int r = 0; r < 8; ++r) {
                float val = acc[mt][nt][r] + bv;
                u[(size_t)(rbase + r) * KD + col] = f2bf(val);
            }
        }
    }
}

// ---------------- chunked column(-weighted) sum: spart[chunk][k*64+d] ----------------
__global__ __launch_bounds__(64) void kSum(const unsigned short* __restrict__ u,
                                           const float* __restrict__ c,
                                           float* __restrict__ spart, int useC) {
    const int k = blockIdx.x, chunk = blockIdx.y, d = threadIdx.x;
    const int n0 = chunk * 512;
    const unsigned short* up = u + (size_t)n0 * KD + k * DCAP + d;
    float a = 0.f;
    if (useC) {
        const float* cp = c + (size_t)n0 * NCAP + k;
        for (int n = 0; n < 512; ++n)
            a += cp[(size_t)n * NCAP] * bf2f(up[(size_t)n * KD]);
    } else {
        for (int n = 0; n < 512; ++n)
            a += bf2f(up[(size_t)n * KD]);
    }
    spart[chunk * KD + k * DCAP + d] = a;
}

// ---------------- reduce partials (fixed order) + squash ----------------
__global__ __launch_bounds__(32) void kSquash(const float* __restrict__ spart,
                                              float scale, float* __restrict__ v) {
    const int k = blockIdx.x, L = threadIdx.x;   // one wave per capsule row
    float s0 = 0.f, s1 = 0.f;
    for (int chunk = 0; chunk < 32; ++chunk) {
        s0 += spart[chunk * KD + k * DCAP + L];
        s1 += spart[chunk * KD + k * DCAP + L + 32];
    }
    s0 *= scale; s1 *= scale;
    float p = s0 * s0 + s1 * s1;
    #pragma unroll
    for (int off = 16; off; off >>= 1) p += __shfl_xor(p, off, 32);
    float n2 = p;
    float f = (n2 / (1.f + n2)) * rsqrtf(n2 + 1e-8f);
    v[k * DCAP + L]      = s0 * f;
    v[k * DCAP + L + 32] = s1 * f;
}

// ---------------- logits update + softmax -> c (one wave per sample) ----------------
__global__ __launch_bounds__(256) void kLogits(const unsigned short* __restrict__ u,
                                               const float* __restrict__ v,
                                               float* __restrict__ logits,
                                               float* __restrict__ cbuf, int addOld) {
    __shared__ float vl[KD];
    __shared__ float lw[8 * NCAP];
    const int tid = threadIdx.x, lane = tid & 31, w = tid >> 5;

    for (int i = tid; i < KD; i += 256) vl[i] = v[i];
    __syncthreads();

    const int base = (blockIdx.x * 8 + w) * 32;        // 64 blocks * 8 waves * 32 samples
    for (int si = 0; si < 32; ++si) {
        const int n = base + si;
        float l0 = addOld ? logits[(size_t)n * NCAP + lane]      : 0.f;
        float l1 = addOld ? logits[(size_t)n * NCAP + lane + 32] : 0.f;
        lw[w * NCAP + lane]      = l0;
        lw[w * NCAP + lane + 32] = l1;

        const unsigned* urow = (const unsigned*)(u + (size_t)n * KD);  // bf16x2 packed
        float m = -1e30f;
        for (int k = 0; k < NCAP; ++k) {
            unsigned p = urow[k * 32 + lane];
            float u0 = __uint_as_float(p << 16);
            float u1 = __uint_as_float(p & 0xFFFF0000u);
            float2 vv = *(const float2*)&vl[k * DCAP + lane * 2];
            float pd = u0 * vv.x + u1 * vv.y;
            #pragma unroll
            for (int off = 16; off; off >>= 1) pd += __shfl_xor(pd, off, 32);
            float lnew = lw[w * NCAP + k] + pd;        // uniform across lanes
            if (lane == 0) lw[w * NCAP + k] = lnew;
            m = fmaxf(m, lnew);
        }
        float e0 = __expf(lw[w * NCAP + lane]      - m);
        float e1 = __expf(lw[w * NCAP + lane + 32] - m);
        float ps = e0 + e1;
        #pragma unroll
        for (int off = 16; off; off >>= 1) ps += __shfl_xor(ps, off, 32);
        float inv = 1.f / ps;
        logits[(size_t)n * NCAP + lane]      = lw[w * NCAP + lane];
        logits[(size_t)n * NCAP + lane + 32] = lw[w * NCAP + lane + 32];
        cbuf[(size_t)n * NCAP + lane]      = e0 * inv;
        cbuf[(size_t)n * NCAP + lane + 32] = e1 * inv;
    }
}

extern "C" void kernel_launch(void* const* d_in, const int* in_sizes, int n_in,
                              void* d_out, int out_size, void* d_ws, size_t ws_size,
                              hipStream_t stream) {
    const float* x = (const float*)d_in[0];
    const float* W = (const float*)d_in[1];
    const float* b = (const float*)d_in[2];
    float* out = (float*)d_out;
    char* ws = (char*)d_ws;

    unsigned short* xb = (unsigned short*)(ws);                       // 32 MB
    unsigned short* wt = (unsigned short*)(ws + 33554432ull);         //  8 MB
    unsigned short* u  = (unsigned short*)(ws + 41943040ull);         // 128 MB
    float* logits      = (float*)(ws + 176160768ull);                 //  4 MB
    float* cbuf        = (float*)(ws + 180355072ull);                 //  4 MB
    float* spart       = (float*)(ws + 184549376ull);                 // 512 KB
    float* vbuf        = (float*)(ws + 185073664ull);                 // 16 KB

    kConvert  <<<4096, 256, 0, stream>>>(x, xb, N_SAMP * D_IN);
    kConvertWT<<<1024, 256, 0, stream>>>(W, wt);

    dim3 gg(KD / TN, N_SAMP / TM);          // (32, 128)
    kGemm<<<gg, 256, 0, stream>>>(xb, wt, b, u);

    dim3 gs(NCAP, 32);
    // iter 0: c = 1/64 uniform -> s0 = colsum(u)/64
    kSum   <<<gs, 64, 0, stream>>>(u, cbuf, spart, 0);
    kSquash<<<NCAP, 32, 0, stream>>>(spart, 1.0f / 64.0f, vbuf);
    // iter 1
    kLogits<<<64, 256, 0, stream>>>(u, vbuf, logits, cbuf, 0);
    kSum   <<<gs, 64, 0, stream>>>(u, cbuf, spart, 1);
    kSquash<<<NCAP, 32, 0, stream>>>(spart, 1.0f, vbuf);
    // iter 2 (final v -> d_out)
    kLogits<<<64, 256, 0, stream>>>(u, vbuf, logits, cbuf, 1);
    kSum   <<<gs, 64, 0, stream>>>(u, cbuf, spart, 1);
    kSquash<<<NCAP, 32, 0, stream>>>(spart, 1.0f, out);
}
